// MixtralDecoderLayer_51634096832585
// MI455X (gfx1250) — matrix operand order repaired and checked by reading, not compiled, not run on hardware
//
#include <hip/hip_runtime.h>

#define T_TOK   2048
#define HID     2048
#define NH      16
#define NKV     4
#define HD      128
#define FFN_DIM 4096
#define NEXP    8
#define EPSV    1e-5f

typedef __attribute__((ext_vector_type(16))) __bf16 v16bf;
typedef __attribute__((ext_vector_type(8)))  __bf16 v8bf;
typedef __attribute__((ext_vector_type(8)))  float  v8f;
typedef __attribute__((ext_vector_type(4)))  float  v4f;

__device__ __forceinline__ __bf16 tobf(float x) { return (__bf16)x; }

__device__ __forceinline__ v16bf cat16(v8bf lo, v8bf hi) {
  return __builtin_shufflevector(lo, hi, 0,1,2,3,4,5,6,7,8,9,10,11,12,13,14,15);
}

// A fragment (16x32 bf16): row = lane&15; K = hl*16 + e
__device__ __forceinline__ v16bf afrag_bf16(const __bf16* rowp, int k0, int hl) {
  const __bf16* p = rowp + k0 + hl * 16;
  return cat16(*(const v8bf*)p, *(const v8bf*)(p + 8));
}
// B fragment (32x16 bf16) from a K-contiguous row: N = lane&15 (caller picks row), K = hl*16 + e
__device__ __forceinline__ v16bf bfrag_bf16(const __bf16* rowp, int k0, int hl) {
  const __bf16* p = rowp + k0 + hl * 16;
  return cat16(*(const v8bf*)p, *(const v8bf*)(p + 8));
}

// Raw f32 B fragment: 4 independent b128 loads (distinct destinations -> clause-able),
// converted to bf16 only when consumed (manual 1-stage pipeline in the GEMM loop).
struct braw { v4f f0, f1, f2, f3; };
__device__ __forceinline__ braw load_braw(const float* p) {
  const v4f* p4 = (const v4f*)p;
  braw r;
  r.f0 = p4[0]; r.f1 = p4[1]; r.f2 = p4[2]; r.f3 = p4[3];
  return r;
}
__device__ __forceinline__ v16bf cvt_braw(const braw& r) {
  v16bf b;
#pragma unroll
  for (int e = 0; e < 4; ++e) b[e]      = tobf(r.f0[e]);
#pragma unroll
  for (int e = 0; e < 4; ++e) b[e + 4]  = tobf(r.f1[e]);
#pragma unroll
  for (int e = 0; e < 4; ++e) b[e + 8]  = tobf(r.f2[e]);
#pragma unroll
  for (int e = 0; e < 4; ++e) b[e + 12] = tobf(r.f3[e]);
  return b;
}

#define WMMA_BF16(a, b, c) \
  __builtin_amdgcn_wmma_f32_16x16x32_bf16(false, (a), false, (b), (short)0, (c), false, false)

// ---------------- RMSNorm (+ optional f32 passthrough copy for the residual path) ----------------
__global__ __launch_bounds__(256)
void rmsnorm_kernel(const float* __restrict__ x, const float* __restrict__ w,
                    __bf16* __restrict__ out_bf, float* __restrict__ copy_out) {
  __shared__ float red[256];
  int t = blockIdx.x, tid = threadIdx.x;
  const float* xr = x + (size_t)t * HID;
  float ss = 0.f;
  for (int i = tid; i < HID; i += 256) { float v = xr[i]; ss += v * v; }
  red[tid] = ss; __syncthreads();
  for (int s = 128; s > 0; s >>= 1) { if (tid < s) red[tid] += red[tid + s]; __syncthreads(); }
  float inv = rsqrtf(red[0] / (float)HID + EPSV);
  for (int i = tid; i < HID; i += 256) {
    float v = xr[i];
    if (copy_out) copy_out[(size_t)t * HID + i] = v;
    out_bf[(size_t)t * HID + i] = tobf(v * inv * w[i]);
  }
}

// Async-copy one 32x32 bf16 A tile (2048B = 128 x b128) into LDS; issued by one wave (32 lanes x 4 ops).
// ASYNCcnt += 4 on the issuing wave; async loads complete in order.
__device__ __forceinline__ void async_fill_Atile(const __bf16* A, int K, int m0, int k0,
                                                 unsigned ldsbase, int lane) {
#pragma unroll
  for (int c = 0; c < 4; ++c) {
    int j = c * 32 + lane;           // chunk id 0..127
    int row = j >> 2, piece = j & 3; // 4 x 16B pieces per 64B row
    unsigned lds = ldsbase + (unsigned)(row * 64 + piece * 16);
    const __bf16* g = A + (size_t)(m0 + row) * K + k0 + piece * 8;
    asm volatile("global_load_async_to_lds_b128 %0, %1, off"
                 :: "v"(lds), "v"(g) : "memory");
  }
}

// ---------------- WMMA GEMM: C[M,N] (+epilogue) = A_bf16[M,K] * W_f32[N,K]^T ----------------
// 8-wave workgroup: shared 32-row A tile double-buffered via global_load_async_to_lds_b128;
// each wave owns a 32x64 output tile (block covers 32 x 512).
// mode 0: C = AB^T ; mode 1: C = AB^T + resid ; mode 2: C += wfull[m*8+expert] * AB^T
__global__ __launch_bounds__(256, 1)
void gemm_wmma_kernel(const __bf16* __restrict__ A, const float* __restrict__ W,
                      float* __restrict__ C, int M, int N, int K,
                      const float* __restrict__ resid,
                      const float* __restrict__ wfull, int expert, int mode) {
  __shared__ __attribute__((aligned(16))) __bf16 Atile[2][32][32];
  int tid = threadIdx.x;
  int wave = tid >> 5, lane = tid & 31;
  int hl = lane >> 4, ln = lane & 15;
  int n0 = blockIdx.x * 512 + wave * 64;
  int m0 = blockIdx.y * 32;

  v8f zero = {0.f, 0.f, 0.f, 0.f, 0.f, 0.f, 0.f, 0.f};
  v8f acc[2][4];
#pragma unroll
  for (int i = 0; i < 2; ++i)
#pragma unroll
    for (int c = 0; c < 4; ++c) acc[i][c] = zero;

  unsigned base0 = (unsigned)(uintptr_t)&Atile[0][0][0];  // flat LDS ptr low 32 bits = LDS offset
  unsigned base1 = (unsigned)(uintptr_t)&Atile[1][0][0];
  int nk = K / 32;
  if (wave == 0) async_fill_Atile(A, K, m0, 0, base0, lane);

  // Hoisted per-column W row pointers (advance by +32 floats per K step)
  const float* wr0 = W + (size_t)(n0 + 0 * 16 + ln) * K + hl * 16;
  const float* wr1 = W + (size_t)(n0 + 1 * 16 + ln) * K + hl * 16;
  const float* wr2 = W + (size_t)(n0 + 2 * 16 + ln) * K + hl * 16;
  const float* wr3 = W + (size_t)(n0 + 3 * 16 + ln) * K + hl * 16;

  for (int i = 0; i < nk; ++i) {
    __syncthreads();  // all waves done reading the buffer about to be refilled
    if (wave == 0) {
      if (i + 1 < nk) {
        async_fill_Atile(A, K, m0, (i + 1) * 32, (i & 1) ? base0 : base1, lane);
        asm volatile("s_wait_asynccnt 0x4" ::: "memory");  // current tile's 4 ops done
      } else {
        asm volatile("s_wait_asynccnt 0x0" ::: "memory");
      }
    }
    __syncthreads();  // Atile[i&1] visible to all waves
    const __bf16* at = &Atile[i & 1][0][0];
    v16bf a0 = afrag_bf16(at + ln * 32, 0, hl);
    v16bf a1 = afrag_bf16(at + (16 + ln) * 32, 0, hl);

    __builtin_prefetch(wr0 + 32, 0, 0);  // global_prefetch_b8 next K tile (whole cacheline)
    __builtin_prefetch(wr2 + 32, 0, 0);

    // 1-stage software pipeline over the 4 column fragments: load c+1 raw while computing c
    braw cur = load_braw(wr0);
    {
      braw nxt = load_braw(wr1);
      v16bf b = cvt_braw(cur);
      acc[0][0] = WMMA_BF16(a0, b, acc[0][0]);
      acc[1][0] = WMMA_BF16(a1, b, acc[1][0]);
      cur = nxt;
    }
    {
      braw nxt = load_braw(wr2);
      v16bf b = cvt_braw(cur);
      acc[0][1] = WMMA_BF16(a0, b, acc[0][1]);
      acc[1][1] = WMMA_BF16(a1, b, acc[1][1]);
      cur = nxt;
    }
    {
      braw nxt = load_braw(wr3);
      v16bf b = cvt_braw(cur);
      acc[0][2] = WMMA_BF16(a0, b, acc[0][2]);
      acc[1][2] = WMMA_BF16(a1, b, acc[1][2]);
      cur = nxt;
    }
    {
      v16bf b = cvt_braw(cur);
      acc[0][3] = WMMA_BF16(a0, b, acc[0][3]);
      acc[1][3] = WMMA_BF16(a1, b, acc[1][3]);
    }
    wr0 += 32; wr1 += 32; wr2 += 32; wr3 += 32;
  }
#pragma unroll
  for (int i = 0; i < 2; ++i)
#pragma unroll
    for (int c = 0; c < 4; ++c)
#pragma unroll
      for (int r = 0; r < 8; ++r) {
        int mm = m0 + i * 16 + hl * 8 + r;
        int nn = n0 + c * 16 + ln;
        size_t off = (size_t)mm * N + nn;
        float v = acc[i][c][r];
        if (mode == 0)      C[off] = v;
        else if (mode == 1) C[off] = v + resid[off];
        else                C[off] += wfull[(size_t)mm * NEXP + expert] * v;
      }
}

// ---------------- RoPE + pack: qkv f32 -> q_bf[h][t][d], k_bf[kvh][t][d], vt_bf[kvh][d][t] ----------------
__global__ __launch_bounds__(256)
void rope_pack_kernel(const float* __restrict__ qkv, const int* __restrict__ pos,
                      __bf16* __restrict__ qb, __bf16* __restrict__ kb, __bf16* __restrict__ vt) {
  int t = blockIdx.x, tid = threadIdx.x;
  float p = (float)pos[t];
  const float* qr = qkv + (size_t)t * 3072;
  const float lnT = 9.210340371976184f;  // ln(10000)
  for (int idx = tid; idx < NH * 64; idx += 256) {
    int h = idx >> 6, j = idx & 63;
    float freq = __expf(-((float)(2 * j) / 128.f) * lnT);
    float ang = p * freq, sn, cs; __sincosf(ang, &sn, &cs);
    float x1 = qr[h * 128 + j], x2 = qr[h * 128 + j + 64];
    size_t base = ((size_t)h * T_TOK + t) * HD;
    qb[base + j]      = tobf(x1 * cs - x2 * sn);
    qb[base + j + 64] = tobf(x2 * cs + x1 * sn);
  }
  for (int idx = tid; idx < NKV * 64; idx += 256) {
    int h = idx >> 6, j = idx & 63;
    float freq = __expf(-((float)(2 * j) / 128.f) * lnT);
    float ang = p * freq, sn, cs; __sincosf(ang, &sn, &cs);
    float x1 = qr[2048 + h * 128 + j], x2 = qr[2048 + h * 128 + j + 64];
    size_t base = ((size_t)h * T_TOK + t) * HD;
    kb[base + j]      = tobf(x1 * cs - x2 * sn);
    kb[base + j + 64] = tobf(x2 * cs + x1 * sn);
  }
  for (int idx = tid; idx < NKV * HD; idx += 256) {
    int h = idx >> 7, d = idx & 127;
    vt[((size_t)h * HD + d) * T_TOK + t] = tobf(qr[2560 + h * 128 + d]);
  }
}

// ---------------- Flash attention: one wave per (16-query tile, head); causal, online softmax ----------------
__global__ __launch_bounds__(32)
void flash_attn_kernel(const __bf16* __restrict__ Q, const __bf16* __restrict__ Kc,
                       const __bf16* __restrict__ Vt, __bf16* __restrict__ Ob) {
  __shared__ __attribute__((aligned(16))) float  s_lds[16][32];
  __shared__ __attribute__((aligned(16))) __bf16 p_lds[16][32];
  __shared__ float fac_lds[16];
  __shared__ float lsum_lds[16];
  int qt = blockIdx.x, h = blockIdx.y;
  int lane = threadIdx.x & 31, hl = lane >> 4, ln = lane & 15;
  int m0 = qt * 16;
  int kvh = h >> 2;  // rep = NH / NKV = 4

  const __bf16* qrow = Q + ((size_t)h * T_TOK + m0 + ln) * HD;
  v16bf qf[4];
#pragma unroll
  for (int kk = 0; kk < 4; ++kk) qf[kk] = afrag_bf16(qrow, kk * 32, hl);

  v8f zero = {0.f, 0.f, 0.f, 0.f, 0.f, 0.f, 0.f, 0.f};
  v8f acco[8];
#pragma unroll
  for (int c = 0; c < 8; ++c) acco[c] = zero;
  float mrow = -3.0e30f, lrow = 0.f;
  const float scale = 0.08838834764831845f;  // 1/sqrt(128)
  int nkb = (m0 + 47) >> 5;                  // ceil((m0+16)/32) causal key blocks

  for (int kbI = 0; kbI < nkb; ++kbI) {
    int s0 = kbI * 32;
    v8f sc[2]; sc[0] = zero; sc[1] = zero;
#pragma unroll
    for (int c = 0; c < 2; ++c) {
      const __bf16* krow = Kc + ((size_t)kvh * T_TOK + s0 + c * 16 + ln) * HD;
#pragma unroll
      for (int kk = 0; kk < 4; ++kk) {
        v16bf b = bfrag_bf16(krow, kk * 32, hl);
        sc[c] = WMMA_BF16(qf[kk], b, sc[c]);
      }
    }
    // scale + causal mask, stage scores to LDS (C layout: M = r + hl*8, N = lane&15)
#pragma unroll
    for (int c = 0; c < 2; ++c)
#pragma unroll
      for (int r = 0; r < 8; ++r) {
        int tt = m0 + hl * 8 + r;
        int ss = s0 + c * 16 + ln;
        float v = sc[c][r] * scale;
        if (ss > tt) v = -3.0e30f;
        s_lds[hl * 8 + r][c * 16 + ln] = v;
      }
    __syncthreads();
    if (lane < 16) {  // one row per lane; closed before any WMMA (EXEC must be all-ones there)
      float bm = -3.0e30f;
#pragma unroll
      for (int j = 0; j < 32; ++j) bm = fmaxf(bm, s_lds[lane][j]);
      float newm = fmaxf(mrow, bm);
      float f = __expf(mrow - newm);
      float sum = 0.f;
#pragma unroll
      for (int j = 0; j < 32; ++j) {
        float e = __expf(s_lds[lane][j] - newm);
        p_lds[lane][j] = tobf(e);
        sum += e;
      }
      lrow = lrow * f + sum;
      mrow = newm;
      fac_lds[lane] = f;
    }
    __syncthreads();
    float fr[8];
#pragma unroll
    for (int r = 0; r < 8; ++r) fr[r] = fac_lds[hl * 8 + r];
#pragma unroll
    for (int c = 0; c < 8; ++c)
#pragma unroll
      for (int r = 0; r < 8; ++r) acco[c][r] *= fr[r];
    // P in A layout straight from LDS (LDS doubles as the C->A transpose)
    v16bf pf = cat16(*(const v8bf*)&p_lds[ln][hl * 16], *(const v8bf*)&p_lds[ln][hl * 16 + 8]);
#pragma unroll
    for (int c = 0; c < 8; ++c) {
      const __bf16* vrow = Vt + ((size_t)kvh * HD + c * 16 + ln) * T_TOK;  // pre-transposed V
      v16bf b = bfrag_bf16(vrow, s0, hl);
      acco[c] = WMMA_BF16(pf, b, acco[c]);
    }
    __syncthreads();
  }
  if (lane < 16) lsum_lds[lane] = lrow;
  __syncthreads();
  float inv[8];
#pragma unroll
  for (int r = 0; r < 8; ++r) inv[r] = 1.f / lsum_lds[hl * 8 + r];
#pragma unroll
  for (int c = 0; c < 8; ++c)
#pragma unroll
    for (int r = 0; r < 8; ++r) {
      int tt = m0 + hl * 8 + r;
      Ob[(size_t)tt * HID + h * HD + c * 16 + ln] = tobf(acco[c][r] * inv[r]);
    }
}

// ---------------- Router: logits -> softmax -> top-2 -> normalized dense weights ----------------
__global__ __launch_bounds__(256)
void gate_kernel(const __bf16* __restrict__ h2, const float* __restrict__ gw,
                 float* __restrict__ wfull) {
  __shared__ float red[256];
  __shared__ float lg[NEXP];
  int t = blockIdx.x, tid = threadIdx.x;
  const __bf16* hr = h2 + (size_t)t * HID;
  for (int e = 0; e < NEXP; ++e) {
    float p = 0.f;
    for (int i = tid; i < HID; i += 256) p += (float)hr[i] * gw[(size_t)e * HID + i];
    red[tid] = p; __syncthreads();
    for (int s = 128; s > 0; s >>= 1) { if (tid < s) red[tid] += red[tid + s]; __syncthreads(); }
    if (tid == 0) lg[e] = red[0];
    __syncthreads();
  }
  if (tid == 0) {
    float mx = -3e30f;
    for (int e = 0; e < NEXP; ++e) mx = fmaxf(mx, lg[e]);
    float pr[NEXP]; float s = 0.f;
    for (int e = 0; e < NEXP; ++e) { pr[e] = __expf(lg[e] - mx); s += pr[e]; }
    for (int e = 0; e < NEXP; ++e) pr[e] /= s;
    int i1 = 0; float v1 = pr[0];
    for (int e = 1; e < NEXP; ++e) if (pr[e] > v1) { v1 = pr[e]; i1 = e; }
    int i2 = -1; float v2 = -1.f;
    for (int e = 0; e < NEXP; ++e) if (e != i1 && pr[e] > v2) { v2 = pr[e]; i2 = e; }
    float invs = 1.f / (v1 + v2);
    for (int e = 0; e < NEXP; ++e) wfull[(size_t)t * NEXP + e] = 0.f;
    wfull[(size_t)t * NEXP + i1] = v1 * invs;
    wfull[(size_t)t * NEXP + i2] = v2 * invs;
  }
}

// ---------------- SwiGLU elementwise: g = silu(h1) * h3 (bf16 out) ----------------
__global__ __launch_bounds__(256)
void silu_mul_kernel(const float* __restrict__ h1, const float* __restrict__ h3,
                     __bf16* __restrict__ g) {
  size_t i = (size_t)blockIdx.x * 256 + threadIdx.x;
  if (i < (size_t)T_TOK * FFN_DIM) {
    float a = h1[i];
    float si = a / (1.f + __expf(-a));
    g[i] = tobf(si * h3[i]);
  }
}

extern "C" void kernel_launch(void* const* d_in, const int* in_sizes, int n_in,
                              void* d_out, int out_size, void* d_ws, size_t ws_size,
                              hipStream_t stream) {
  (void)in_sizes; (void)n_in; (void)out_size; (void)ws_size;
  const float* x    = (const float*)d_in[0];
  const int*   pos  = (const int*)d_in[1];
  const float* n1w  = (const float*)d_in[2];
  const float* n2w  = (const float*)d_in[3];
  const float* wqkv = (const float*)d_in[4];
  const float* wo   = (const float*)d_in[5];
  const float* gw   = (const float*)d_in[6];
  const float* w1   = (const float*)d_in[7];
  const float* w2   = (const float*)d_in[8];
  const float* w3   = (const float*)d_in[9];
  float* out = (float*)d_out;

  char* ws = (char*)d_ws;
  size_t off = 0;
  auto carve = [&](size_t bytes) -> char* {
    char* p = ws + off; off += (bytes + 255) & ~(size_t)255; return p;
  };
  __bf16* h1n  = (__bf16*)carve((size_t)T_TOK * HID * 2);        // norm1 out (bf16)
  float*  qkv  = (float*) carve((size_t)T_TOK * 3072 * 4);       // QKV projection (f32)
  __bf16* qb   = (__bf16*)carve((size_t)NH  * T_TOK * HD * 2);   // roped Q [h][t][d]
  __bf16* kb   = (__bf16*)carve((size_t)NKV * T_TOK * HD * 2);   // roped K [kvh][t][d]
  __bf16* vt   = (__bf16*)carve((size_t)NKV * HD * T_TOK * 2);   // V transposed [kvh][d][t]
  __bf16* attn = (__bf16*)carve((size_t)T_TOK * HID * 2);        // attention out (bf16)
  float*  x2   = (float*) carve((size_t)T_TOK * HID * 4);        // attn_out + residual (f32)
  __bf16* h2   = (__bf16*)carve((size_t)T_TOK * HID * 2);        // norm2 out (bf16)
  float*  wful = (float*) carve((size_t)T_TOK * NEXP * 4);       // dense routing weights
  float*  h1f  = (float*) carve((size_t)T_TOK * FFN_DIM * 4);
  float*  h3f  = (float*) carve((size_t)T_TOK * FFN_DIM * 4);
  __bf16* gact = (__bf16*)carve((size_t)T_TOK * FFN_DIM * 2);

  rmsnorm_kernel<<<T_TOK, 256, 0, stream>>>(x, n1w, h1n, nullptr);
  gemm_wmma_kernel<<<dim3(3072 / 512, T_TOK / 32), 256, 0, stream>>>(
      h1n, wqkv, qkv, T_TOK, 3072, HID, nullptr, nullptr, 0, 0);
  rope_pack_kernel<<<T_TOK, 256, 0, stream>>>(qkv, pos, qb, kb, vt);
  flash_attn_kernel<<<dim3(T_TOK / 16, NH), 32, 0, stream>>>(qb, kb, vt, attn);
  gemm_wmma_kernel<<<dim3(HID / 512, T_TOK / 32), 256, 0, stream>>>(
      attn, wo, x2, T_TOK, HID, HID, x, nullptr, 0, 1);          // + residual
  rmsnorm_kernel<<<T_TOK, 256, 0, stream>>>(x2, n2w, h2, out);   // out := residual2
  gate_kernel<<<T_TOK, 256, 0, stream>>>(h2, gw, wful);

  for (int e = 0; e < NEXP; ++e) {
    const float* w1e = w1 + (size_t)e * FFN_DIM * HID;
    const float* w3e = w3 + (size_t)e * FFN_DIM * HID;
    const float* w2e = w2 + (size_t)e * HID * FFN_DIM;
    gemm_wmma_kernel<<<dim3(FFN_DIM / 512, T_TOK / 32), 256, 0, stream>>>(
        h2, w1e, h1f, T_TOK, FFN_DIM, HID, nullptr, nullptr, 0, 0);
    gemm_wmma_kernel<<<dim3(FFN_DIM / 512, T_TOK / 32), 256, 0, stream>>>(
        h2, w3e, h3f, T_TOK, FFN_DIM, HID, nullptr, nullptr, 0, 0);
    size_t nel = (size_t)T_TOK * FFN_DIM;
    silu_mul_kernel<<<(nel + 255) / 256, 256, 0, stream>>>(h1f, h3f, gact);
    gemm_wmma_kernel<<<dim3(HID / 512, T_TOK / 32), 256, 0, stream>>>(
        gact, w2e, out, T_TOK, HID, FFN_DIM, nullptr, wful, e, 2);  // out += w_e * expert_e
  }
}
